// Attention_2173253452559
// MI455X (gfx1250) — matrix operand order repaired and checked by reading, not compiled, not run on hardware
//
#include <hip/hip_runtime.h>

// ---------------------------------------------------------------------------
// MI455X (gfx1250) multi-head attention: QKV GEMM -> flash attention -> proj
// bf16 WMMA (v_wmma_f32_16x16x32_bf16), async global->LDS staging (ASYNCcnt),
// softmax row-sums folded into an extra WMMA tile, VALU permlane reductions.
// Workspace requirement: ~88 MB (layout in kernel_launch).
// ---------------------------------------------------------------------------

typedef __attribute__((ext_vector_type(16))) __bf16 v16bf;
typedef __attribute__((ext_vector_type(8)))  float  v8f;

#define WMMA_BF16(a, b, c) \
  __builtin_amdgcn_wmma_f32_16x16x32_bf16(false, (a), false, (b), (short)0, (c), false, false)

constexpr int   Bb   = 4;
constexpr int   Nn   = 2048;
constexpr int   Cc   = 1024;
constexpr int   Hh   = 16;
constexpr int   Dd   = 64;
constexpr int   ROWS = Bb * Nn;     // 8192
constexpr int   QKVC = 3 * Cc;      // 3072
constexpr int   NCH  = Nn / 32;     // 64 key chunks
constexpr float SM_SCALE = 0.125f;  // 1/sqrt(64), exact power of two in bf16

// CDNA5 async global->LDS copy (ASYNCcnt path), 16 bytes per lane.
__device__ __forceinline__ void async_ld_b128_to_lds(void* lds_dst, const void* gsrc) {
  unsigned off = (unsigned)(uintptr_t)lds_dst;   // low 32 bits = LDS offset
  asm volatile("global_load_async_to_lds_b128 %0, %1, off"
               :: "v"(off), "v"(gsrc) : "memory");
}
__device__ __forceinline__ void wait_async0() {
  asm volatile("s_wait_asynccnt 0x0" ::: "memory");
}

// 16-lane XOR-butterfly max, pure VALU via v_permlane16_b32 (no DScnt waits).
__device__ __forceinline__ float red_max16(float v) {
#if __has_builtin(__builtin_amdgcn_permlane16)
  #define PLX(s0, s1)                                                     \
    v = fmaxf(v, __int_as_float(__builtin_amdgcn_permlane16(              \
            __float_as_int(v), __float_as_int(v), (s0), (s1), false, false)))
  PLX(0x67452301, 0xEFCDAB89);   // xor 1
  PLX(0x54761032, 0xDCFE98BA);   // xor 2
  PLX(0x32107654, 0xBA98FEDC);   // xor 4
  PLX(0xFEDCBA98, 0x76543210);   // xor 8
  #undef PLX
#else
  v = fmaxf(v, __shfl_xor(v, 1, 32));
  v = fmaxf(v, __shfl_xor(v, 2, 32));
  v = fmaxf(v, __shfl_xor(v, 4, 32));
  v = fmaxf(v, __shfl_xor(v, 8, 32));
#endif
  return v;
}

// ---------------------------------------------------------------------------
__global__ void cvt_f32_to_bf16(const float* __restrict__ in,
                                __bf16* __restrict__ out, int n) {
  int i = blockIdx.x * blockDim.x + threadIdx.x;
  if (i < n) out[i] = (__bf16)in[i];
}

// ---------------------------------------------------------------------------
// C[M x Nc] = A[M x K] @ W^T + bias   (W: [Nc x K], torch Linear layout)
// One wave computes a 64x64 tile: 16 WMMA accumulators; each B fragment is
// reused by 4 WMMAs (4 M-subtiles).
// A-frag: lane(0..15)=row, K(e) = e + 16*hi
// B-frag: lane(0..15)=col, K(e) = e + 16*hi
// C/D   : row = r + 8*hi, col = lane&15
// ---------------------------------------------------------------------------
template <bool STORE_BF16>
__global__ void gemm_bf16_wmma(const __bf16* __restrict__ A,
                               const __bf16* __restrict__ W,
                               const float* __restrict__ bias,
                               void* __restrict__ Cout,
                               int M, int K, int Nc) {
  const int lane = threadIdx.x & 31;
  const int wave = blockIdx.x * (blockDim.x >> 5) + (threadIdx.x >> 5);
  const int ntiles = Nc >> 6;
  const int mtile = wave / ntiles;
  const int ntile = wave - mtile * ntiles;
  if (mtile * 64 >= M) return;
  const int m0 = mtile * 64, n0 = ntile * 64;
  const int col = lane & 15, hi = lane >> 4;

  v8f acc[4][4] = {};                       // [n-subtile][m-subtile]
  for (int k0 = 0; k0 < K; k0 += 32) {
    v16bf a[4];
#pragma unroll
    for (int s = 0; s < 4; ++s) {
      const __bf16* ap = A + (size_t)(m0 + s * 16 + col) * K + k0 + hi * 16;
      __builtin_prefetch(ap + 64, 0, 0);
#pragma unroll
      for (int e = 0; e < 8; ++e) { a[s][e] = ap[e]; a[s][e + 8] = ap[e + 8]; }
    }
#pragma unroll
    for (int t = 0; t < 4; ++t) {
      const __bf16* wp = W + (size_t)(n0 + t * 16 + col) * K + k0 + hi * 16;
      v16bf bb;
#pragma unroll
      for (int e = 0; e < 16; ++e) bb[e] = wp[e];
#pragma unroll
      for (int s = 0; s < 4; ++s) acc[t][s] = WMMA_BF16(a[s], bb, acc[t][s]);
    }
  }
#pragma unroll
  for (int t = 0; t < 4; ++t) {
    const int cc = n0 + t * 16 + col;
    const float bv = bias[cc];
#pragma unroll
    for (int s = 0; s < 4; ++s) {
#pragma unroll
      for (int r = 0; r < 8; ++r) {
        const int row = m0 + s * 16 + r + hi * 8;
        const float v = acc[t][s][r] + bv;
        if (STORE_BF16) ((__bf16*)Cout)[(size_t)row * Nc + cc] = (__bf16)v;
        else            ((float*) Cout)[(size_t)row * Nc + cc] = v;
      }
    }
  }
}

// ---------------------------------------------------------------------------
// Flash attention. qkv: bf16 [ROWS][3072] = [B,N,3,H,D].
// Block = 256 thr = 8 waves sharing one (b,h). Each wave owns 16 query rows.
// Keys stream in 32-wide chunks, double-buffered:
//   K chunk -> kbuf  via global_load_async_to_lds_b128
//   V chunk -> vbufT via explicit load + transposed ds stores
// Q is pre-scaled by 1/sqrt(D) = 0.125 (exact in bf16), so scores come out of
// the QK^T WMMA already scaled. The softmax denominator is accumulated as a
// 5th WMMA output tile against a constant ones B-fragment (P @ 1 = row sums),
// so only the row-max needs a cross-lane reduction (v_permlane16_b32).
// ---------------------------------------------------------------------------
__global__ void flash_attn_wmma(const __bf16* __restrict__ qkv,
                                __bf16* __restrict__ attn_out) {
  __shared__ alignas(16) __bf16 kbuf[2][32][72];    // 9.0 KB (row padded)
  __shared__ alignas(16) __bf16 vbufT[2][Dd][40];   // 10.0 KB (transposed V)
  __shared__ alignas(16) __bf16 plds[8][16][40];    // 10.0 KB per-wave P

  const int tid   = threadIdx.x;
  const int lane  = tid & 31;
  const int wslot = tid >> 5;
  const int col   = lane & 15;
  const int hi    = lane >> 4;

  const int wid  = blockIdx.x * 8 + wslot;
  const int qblk = wid & 127;                       // N/16 = 128
  const int bh   = wid >> 7;
  const int b    = bh >> 4;
  const int h    = bh & 15;
  const int m0   = qblk * 16;

  // Q fragments (two 16x32 A fragments over D=64), pre-scaled by 1/sqrt(D).
  // 0.125f is a power of two -> exact in bf16, no precision change.
  v16bf aq[2];
#pragma unroll
  for (int kc = 0; kc < 2; ++kc) {
    const __bf16* qp = qkv + (size_t)(b * Nn + m0 + col) * QKVC +
                       h * Dd + kc * 32 + hi * 16;
#pragma unroll
    for (int e = 0; e < 8; ++e) {
      aq[kc][e]     = (__bf16)((float)qp[e]      * SM_SCALE);
      aq[kc][e + 8] = (__bf16)((float)qp[e + 8] * SM_SCALE);
    }
  }

  // constant ones B-fragment (for P @ 1 row sums)
  v16bf vone;
#pragma unroll
  for (int e = 0; e < 16; ++e) vone[e] = (__bf16)1.0f;

  float mrow[8];
#pragma unroll
  for (int r = 0; r < 8; ++r) mrow[r] = -INFINITY;
  v8f o[5] = {};                                    // o[4] = running row sums

  // staging coordinates: each thread moves one 16B segment of K and of V
  const int srow = tid >> 3;            // key row within chunk: 0..31
  const int scol = (tid & 7) * 8;       // 8 bf16 = 16B
  const size_t headbase = (size_t)(b * Nn) * QKVC + h * Dd;

  auto stage_issue = [&](int kb2, int buf, uint4& vreg) {
    const size_t roff = headbase + (size_t)(kb2 * 32 + srow) * QKVC + scol;
    async_ld_b128_to_lds(&kbuf[buf][srow][scol], qkv + roff + Cc);   // K async
    vreg = *(const uint4*)(qkv + roff + 2 * Cc);                     // V to regs
  };
  auto stage_store_v = [&](int buf, const uint4& vreg) {
    __bf16 ve[8];
    *(uint4*)ve = vreg;
#pragma unroll
    for (int j = 0; j < 8; ++j) vbufT[buf][scol + j][srow] = ve[j];  // transpose
  };

  // prologue: stage chunk 0 into buffer 0
  {
    uint4 v0;
    stage_issue(0, 0, v0);
    stage_store_v(0, v0);
    wait_async0();
  }
  __syncthreads();

  for (int kb = 0; kb < NCH; ++kb) {
    const int cur = kb & 1, nxt = cur ^ 1;
    const bool more = (kb + 1) < NCH;
    uint4 vnext;
    if (more) stage_issue(kb + 1, nxt, vnext);     // overlap with compute

    // ---- S = (Q/8) @ K^T: two 16x16 score tiles over the 32-key chunk ----
    v8f s0 = {}, s1 = {};
#pragma unroll
    for (int kc = 0; kc < 2; ++kc) {
      v16bf bk0, bk1;
      const __bf16* k0p = &kbuf[cur][col][kc * 32 + hi * 16];
      const __bf16* k1p = &kbuf[cur][col + 16][kc * 32 + hi * 16];
#pragma unroll
      for (int e = 0; e < 16; ++e) { bk0[e] = k0p[e]; bk1[e] = k1p[e]; }
      s0 = WMMA_BF16(aq[kc], bk0, s0);
      s1 = WMMA_BF16(aq[kc], bk1, s1);
    }

    // ---- online softmax: row-max via permlane butterfly, exp, rescale ----
#pragma unroll
    for (int r = 0; r < 8; ++r) {
      const float v0 = s0[r];
      const float v1 = s1[r];
      const float mx = red_max16(fmaxf(v0, v1));
      const float mnew = fmaxf(mrow[r], mx);
      const float e0 = __expf(v0 - mnew);
      const float e1 = __expf(v1 - mnew);
      const float lc = __expf(mrow[r] - mnew);
      mrow[r] = mnew;
      plds[wslot][r + hi * 8][col]      = (__bf16)e0;
      plds[wslot][r + hi * 8][col + 16] = (__bf16)e1;
#pragma unroll
      for (int t = 0; t < 5; ++t) o[t][r] *= lc;
    }

    // ---- O += P @ V (4 tiles) and rowsum += P @ 1 (5th tile) ----
    v16bf ap;
    {
      const __bf16* pp = &plds[wslot][col][hi * 16];
#pragma unroll
      for (int e = 0; e < 8; ++e) { ap[e] = pp[e]; ap[e + 8] = pp[e + 8]; }
    }
#pragma unroll
    for (int t = 0; t < 4; ++t) {
      v16bf bv;
      const __bf16* vp = &vbufT[cur][t * 16 + col][hi * 16];
#pragma unroll
      for (int e = 0; e < 16; ++e) bv[e] = vp[e];
      o[t] = WMMA_BF16(ap, bv, o[t]);
    }
    o[4] = WMMA_BF16(ap, vone, o[4]);              // row sums of P

    if (more) stage_store_v(nxt, vnext);
    wait_async0();          // K chunk kb+1 resident before anyone reads it
    __syncthreads();
  }

  // ---- normalize, store to attn buffer [B,N,H,D] = [8192][1024] bf16 ----
#pragma unroll
  for (int r = 0; r < 8; ++r) {
    const float inv = 1.0f / o[4][r];
    const int seq = m0 + r + hi * 8;
#pragma unroll
    for (int t = 0; t < 4; ++t) {
      attn_out[(size_t)(b * Nn + seq) * Cc + h * Dd + t * 16 + col] =
          (__bf16)(o[t][r] * inv);
    }
  }
}

// ---------------------------------------------------------------------------
extern "C" void kernel_launch(void* const* d_in, const int* in_sizes, int n_in,
                              void* d_out, int out_size, void* d_ws, size_t ws_size,
                              hipStream_t stream) {
  (void)in_sizes; (void)n_in; (void)out_size; (void)ws_size;
  const float* x      = (const float*)d_in[0];   // [4,2048,1024]
  const float* w_qkv  = (const float*)d_in[1];   // [3072,1024]
  const float* b_qkv  = (const float*)d_in[2];   // [3072]
  const float* w_proj = (const float*)d_in[3];   // [1024,1024]
  const float* b_proj = (const float*)d_in[4];   // [1024]
  float* out = (float*)d_out;                    // [4,2048,1024]

  // workspace layout (bf16 elements); total ~88 MB
  __bf16* xb     = (__bf16*)d_ws;                         // 8192*1024
  __bf16* wqkvb  = xb    + (size_t)ROWS * Cc;             // 3072*1024
  __bf16* wprojb = wqkvb + (size_t)QKVC * Cc;             // 1024*1024
  __bf16* qkvb   = wprojb + (size_t)Cc * Cc;              // 8192*3072
  __bf16* attnb  = qkvb  + (size_t)ROWS * QKVC;           // 8192*1024

  // 1) fp32 -> bf16
  {
    int n1 = ROWS * Cc, n2 = QKVC * Cc, n3 = Cc * Cc;
    cvt_f32_to_bf16<<<(n1 + 255) / 256, 256, 0, stream>>>(x, xb, n1);
    cvt_f32_to_bf16<<<(n2 + 255) / 256, 256, 0, stream>>>(w_qkv, wqkvb, n2);
    cvt_f32_to_bf16<<<(n3 + 255) / 256, 256, 0, stream>>>(w_proj, wprojb, n3);
  }

  // 2) QKV projection: qkv = x @ w_qkv^T + b_qkv  -> bf16 [8192 x 3072]
  {
    int waves = (ROWS / 64) * (QKVC / 64);     // 128 * 48 = 6144
    gemm_bf16_wmma<true><<<waves / 8, 256, 0, stream>>>(
        xb, wqkvb, b_qkv, qkvb, ROWS, Cc, QKVC);
  }

  // 3) flash attention -> bf16 [8192 x 1024]
  {
    int waves = Bb * Hh * (Nn / 16);           // 8192
    flash_attn_wmma<<<waves / 8, 256, 0, stream>>>(qkvb, attnb);
  }

  // 4) output projection: out = attn @ w_proj^T + b_proj -> fp32
  {
    int waves = (ROWS / 64) * (Cc / 64);       // 128 * 16 = 2048
    gemm_bf16_wmma<false><<<waves / 8, 256, 0, stream>>>(
        attnb, wprojb, b_proj, out, ROWS, Cc, Cc);
  }
}
